// SGNHeadOccLite_31756988187340
// MI455X (gfx1250) — compile-verified, matrix operand order checked by reading, not run on hardware
//
#include <hip/hip_runtime.h>
#include <hip/hip_fp16.h>

typedef __attribute__((ext_vector_type(16))) _Float16 v16h;
typedef __attribute__((ext_vector_type(8)))  _Float16 v8h;
typedef __attribute__((ext_vector_type(8)))  float    v8f;

#define EMBED 128
#define NCAM  2
#define HF    48
#define WF    160
#define HD    370
#define WD    1220
#define NVOX  (128*128*16)

// ---------------------------------------------------------------------------
// Kernel 0: convert weights to transposed f16 (Wt[n][k]) so WMMA B-fragments
// are contiguous 16B loads per lane.
// ---------------------------------------------------------------------------
__global__ void prep_weights(const float* __restrict__ w1,  // (128,128) x@W
                             const float* __restrict__ w2,  // (128,128)
                             const float* __restrict__ w3,  // (128,64)
                             _Float16* __restrict__ W1t,    // [128][128]
                             _Float16* __restrict__ W2t,    // [128][128]
                             _Float16* __restrict__ W3t) {  // [64][128]
  int total = 128*128 + 128*128 + 64*128;
  for (int idx = blockIdx.x * blockDim.x + threadIdx.x; idx < total;
       idx += gridDim.x * blockDim.x) {
    if (idx < 16384) {
      int n = idx >> 7, k = idx & 127;
      W1t[n*128 + k] = (_Float16)w1[k*128 + n];
    } else if (idx < 32768) {
      int t = idx - 16384; int n = t >> 7, k = t & 127;
      W2t[n*128 + k] = (_Float16)w2[k*128 + n];
    } else {
      int t = idx - 32768; int n = t >> 7, k = t & 127;
      W3t[n*128 + k] = (_Float16)w3[k*64 + n];
    }
  }
}

// ---------------------------------------------------------------------------
// Kernel 1: projection + depth/feature sampling + camera pooling + LayerNorm0.
// One block (128 threads, 1 channel each) per voxel. Writes f16 row [vox][128].
// ---------------------------------------------------------------------------
__device__ __forceinline__ float reflectf(float x, float m) {
  x = fabsf(x);
  x = fmodf(x, 2.0f * m);
  return (x > m) ? (2.0f * m - x) : x;
}

__global__ void __launch_bounds__(128)
pool_ln_kernel(const float* __restrict__ feats,   // (2,128,48,160)
               const float* __restrict__ depths,  // (2,370,1220)
               const float* __restrict__ proj,    // (2,4,4)
               const float* __restrict__ g0,
               const float* __restrict__ b0,
               _Float16* __restrict__ A0) {       // [NVOX][128] f16
  int vox = blockIdx.x;
  int ch  = threadIdx.x;
  int i = vox >> 11;            // DX index  (vox / (DY*DZ))
  int j = (vox >> 4) & 127;     // DY index
  int k = vox & 15;             // DZ index
  float x = (i + 0.5f) * 0.4f + 0.0f;
  float y = (j + 0.5f) * 0.4f - 25.6f;
  float z = (k + 0.5f) * 0.4f - 2.0f;

  float acc = 0.f, wcnt = 0.f;
  #pragma unroll
  for (int c = 0; c < NCAM; ++c) {
    const float* M = proj + c * 16;
    float up = M[0]*x + M[1]*y + M[2]*z  + M[3];
    float vp = M[4]*x + M[5]*y + M[6]*z  + M[7];
    float zp = M[8]*x + M[9]*y + M[10]*z + M[11];
    float u = up / zp, v = vp / zp;
    float gu = 2.f * u / (1220.f - 1.f) - 1.f;
    float gv = 2.f * v / (370.f  - 1.f) - 1.f;
    if (__builtin_isinf(gu)) gu = -2.f;
    if (__builtin_isinf(gv)) gv = -2.f;
    bool mask = (fabsf(gu) <= 1.f) && (fabsf(gv) <= 1.f) && (zp > 0.f);

    // nearest depth sample
    float px = ((gu + 1.f) * (float)WD - 1.f) * 0.5f;
    float py = ((gv + 1.f) * (float)HD - 1.f) * 0.5f;
    int ix = (int)rintf(px), iy = (int)rintf(py);
    bool inb = (ix >= 0) && (ix < WD) && (iy >= 0) && (iy < HD);
    int ixc = min(max(ix, 0), WD - 1);
    int iyc = min(max(iy, 0), HD - 1);
    float dsamp = depths[c * HD * WD + iyc * WD + ixc];
    if (!inb) dsamp = 0.f;
    float dz = zp - dsamp;
    float prob = __expf(-dz * dz * (1.0f / 0.64f));

    // bilinear feature sample (reflect padding)
    float fx = reflectf((gu + 1.f) * 0.5f * (float)(WF - 1), (float)(WF - 1));
    float fy = reflectf((gv + 1.f) * 0.5f * (float)(HF - 1), (float)(HF - 1));
    float x0f = floorf(fx), y0f = floorf(fy);
    float wx = fx - x0f, wy = fy - y0f;
    int x0 = min(max((int)x0f, 0), WF - 1);
    int y0 = min(max((int)y0f, 0), HF - 1);
    int x1 = min(x0 + 1, WF - 1);
    int y1 = min(y0 + 1, HF - 1);
    const float* fm = feats + (size_t)(c * EMBED + ch) * HF * WF;
    float s00 = fm[y0 * WF + x0], s01 = fm[y0 * WF + x1];
    float s10 = fm[y1 * WF + x0], s11 = fm[y1 * WF + x1];
    float samp = s00 * (1.f - wy) * (1.f - wx) + s01 * (1.f - wy) * wx
               + s10 * wy * (1.f - wx)        + s11 * wy * wx;
    float featv = samp * prob * 100.f;
    if (mask) { acc += featv; wcnt += 1.f; }
  }
  float w = (wcnt == 0.f) ? 1.f : wcnt;
  float pooled = acc / w;

  // LayerNorm over 128 channels (block reduction)
  __shared__ float red[128];
  __shared__ float red2[128];
  red[ch]  = pooled;
  red2[ch] = pooled * pooled;
  __syncthreads();
  for (int s = 64; s > 0; s >>= 1) {
    if (ch < s) { red[ch] += red[ch + s]; red2[ch] += red2[ch + s]; }
    __syncthreads();
  }
  float mu  = red[0]  * (1.f / 128.f);
  float var = red2[0] * (1.f / 128.f) - mu * mu;
  float o = (pooled - mu) * rsqrtf(var + 1e-5f) * g0[ch] + b0[ch];
  A0[(size_t)vox * 128 + ch] = (_Float16)o;
}

// ---------------------------------------------------------------------------
// WMMA fragment loader: 16x32 f16 tile from a row-major matrix, ldh halves.
// ISA layout: lanes 0-15 -> row M=lane, K {0..7, 16..23}; lanes 16-31 -> row
// M=lane-16, K {8..15, 24..31}. Works for global and LDS pointers.
// ---------------------------------------------------------------------------
template <typename P>
__device__ __forceinline__ v16h load_frag(P base, int ldh, int lane) {
  int r = lane & 15, h = (lane >> 4) & 1;
  P q = base + r * ldh + h * 8;
  v8h lo = *(const v8h*)(q);
  v8h hi = *(const v8h*)(q + 16);
  v16h o;
  #pragma unroll
  for (int i = 0; i < 8; ++i) { o[i] = lo[i]; o[i + 8] = hi[i]; }
  return o;
}

// bias + per-row LayerNorm + leaky-relu, in place on NT accumulator tiles.
// Row m of the 16-row tile lives in VGPR v=(m%8) of lanes (m<8 ? 0-15 : 16-31).
template <int NT>
__device__ __forceinline__ void post_ln(v8f* acc, const float* bias,
                                        const float* g, const float* b,
                                        int lane) {
  int r15 = lane & 15;
  float bn[NT], gn[NT], bbn[NT];
  #pragma unroll
  for (int nt = 0; nt < NT; ++nt) {
    bn[nt]  = bias[nt * 16 + r15];
    gn[nt]  = g[nt * 16 + r15];
    bbn[nt] = b[nt * 16 + r15];
  }
  float psum[8], psq[8];
  #pragma unroll
  for (int v = 0; v < 8; ++v) { psum[v] = 0.f; psq[v] = 0.f; }
  #pragma unroll
  for (int nt = 0; nt < NT; ++nt)
    #pragma unroll
    for (int v = 0; v < 8; ++v) {
      float t = acc[nt][v] + bn[nt];
      acc[nt][v] = t;
      psum[v] += t; psq[v] += t * t;
    }
  #pragma unroll
  for (int m = 1; m < 16; m <<= 1)
    #pragma unroll
    for (int v = 0; v < 8; ++v) {
      psum[v] += __shfl_xor(psum[v], m, 32);
      psq[v]  += __shfl_xor(psq[v],  m, 32);
    }
  const float invN = 1.0f / (float)(NT * 16);
  #pragma unroll
  for (int v = 0; v < 8; ++v) {
    float mu  = psum[v] * invN;
    float var = psq[v] * invN - mu * mu;
    float inv = rsqrtf(var + 1e-5f);
    #pragma unroll
    for (int nt = 0; nt < NT; ++nt) {
      float t = (acc[nt][v] - mu) * inv * gn[nt] + bbn[nt];
      acc[nt][v] = (t > 0.f) ? t : 0.1f * t;
    }
  }
}

// store NT accumulator tiles (D layout) into a 16 x (NT*16) f16 LDS row block.
template <int NT>
__device__ __forceinline__ void store_tile(const v8f* acc, _Float16* xsrow,
                                           int ldout, int lane) {
  int r15 = lane & 15, hh = (lane >> 4) & 1;
  #pragma unroll
  for (int v = 0; v < 8; ++v) {
    int m = v + hh * 8;
    #pragma unroll
    for (int nt = 0; nt < NT; ++nt)
      xsrow[m * ldout + nt * 16 + r15] = (_Float16)acc[nt][v];
  }
}

// ---------------------------------------------------------------------------
// Kernel 2: fused 3-layer WMMA MLP + final 64->1 head. One wave per 16 rows.
// ---------------------------------------------------------------------------
__global__ void __launch_bounds__(256)
mlp_kernel(const _Float16* __restrict__ A0,    // [NVOX][128] f16 (LN0 applied)
           const _Float16* __restrict__ W1t, const float* __restrict__ rb1,
           const float* __restrict__ rg1, const float* __restrict__ rbb1,
           const _Float16* __restrict__ W2t, const float* __restrict__ rb2,
           const float* __restrict__ rg2, const float* __restrict__ rbb2,
           const _Float16* __restrict__ W3t, const float* __restrict__ ob1,
           const float* __restrict__ og, const float* __restrict__ obb,
           const float* __restrict__ w4, const float* __restrict__ b4,
           float* __restrict__ out) {
  int wave = threadIdx.x >> 5;
  int lane = threadIdx.x & 31;
  int tile = blockIdx.x * 8 + wave;
  int row0 = tile * 16;

  __shared__ _Float16 xs[8][16 * 128];   // 4KB per wave staging
  _Float16* myx = xs[wave];

  // ---- Layer 1: x(16x128) @ W1 -> 128, LN, lrelu ----
  v16h a[4];
  #pragma unroll
  for (int kt = 0; kt < 4; ++kt)
    a[kt] = load_frag(A0 + (size_t)row0 * 128 + kt * 32, 128, lane);

  v8f acc[8];
  #pragma unroll
  for (int nt = 0; nt < 8; ++nt) {
    v8f c = {};
    #pragma unroll
    for (int kt = 0; kt < 4; ++kt) {
      v16h bf = load_frag(W1t + nt * 16 * 128 + kt * 32, 128, lane);
      c = __builtin_amdgcn_wmma_f32_16x16x32_f16(false, a[kt], false, bf,
                                                 (short)0, c, false, false);
    }
    acc[nt] = c;
  }
  post_ln<8>(acc, rb1, rg1, rbb1, lane);
  store_tile<8>(acc, myx, 128, lane);
  asm volatile("s_wait_dscnt 0" ::: "memory");

  // ---- Layer 2: 128 -> 128, LN, lrelu ----
  #pragma unroll
  for (int kt = 0; kt < 4; ++kt)
    a[kt] = load_frag(myx + kt * 32, 128, lane);
  #pragma unroll
  for (int nt = 0; nt < 8; ++nt) {
    v8f c = {};
    #pragma unroll
    for (int kt = 0; kt < 4; ++kt) {
      v16h bf = load_frag(W2t + nt * 16 * 128 + kt * 32, 128, lane);
      c = __builtin_amdgcn_wmma_f32_16x16x32_f16(false, a[kt], false, bf,
                                                 (short)0, c, false, false);
    }
    acc[nt] = c;
  }
  post_ln<8>(acc, rb2, rg2, rbb2, lane);
  store_tile<8>(acc, myx, 128, lane);
  asm volatile("s_wait_dscnt 0" ::: "memory");

  // ---- Layer 3: 128 -> 64, LN, lrelu ----
  #pragma unroll
  for (int kt = 0; kt < 4; ++kt)
    a[kt] = load_frag(myx + kt * 32, 128, lane);
  v8f acc3[4];
  #pragma unroll
  for (int nt = 0; nt < 4; ++nt) {
    v8f c = {};
    #pragma unroll
    for (int kt = 0; kt < 4; ++kt) {
      v16h bf = load_frag(W3t + nt * 16 * 128 + kt * 32, 128, lane);
      c = __builtin_amdgcn_wmma_f32_16x16x32_f16(false, a[kt], false, bf,
                                                 (short)0, c, false, false);
    }
    acc3[nt] = c;
  }
  post_ln<4>(acc3, ob1, og, obb, lane);

  // ---- Layer 4: h(16x64) @ w4(64x1) + b4 ----
  int r15 = lane & 15, hh = (lane >> 4) & 1;
  float wn[4];
  #pragma unroll
  for (int nt = 0; nt < 4; ++nt) wn[nt] = w4[nt * 16 + r15];
  float dot[8];
  #pragma unroll
  for (int v = 0; v < 8; ++v) {
    float d = 0.f;
    #pragma unroll
    for (int nt = 0; nt < 4; ++nt) d += acc3[nt][v] * wn[nt];
    dot[v] = d;
  }
  #pragma unroll
  for (int m = 1; m < 16; m <<= 1)
    #pragma unroll
    for (int v = 0; v < 8; ++v) dot[v] += __shfl_xor(dot[v], m, 32);
  if (r15 == 0) {
    float bias = b4[0];
    #pragma unroll
    for (int v = 0; v < 8; ++v)
      out[row0 + hh * 8 + v] = dot[v] + bias;
  }
}

// ---------------------------------------------------------------------------
extern "C" void kernel_launch(void* const* d_in, const int* in_sizes, int n_in,
                              void* d_out, int out_size, void* d_ws, size_t ws_size,
                              hipStream_t stream) {
  const float* feats  = (const float*)d_in[0];
  const float* depths = (const float*)d_in[1];
  const float* proj   = (const float*)d_in[2];
  const float* g0     = (const float*)d_in[3];
  const float* b0     = (const float*)d_in[4];
  const float* rw1    = (const float*)d_in[5];
  const float* rb1    = (const float*)d_in[6];
  const float* rg1    = (const float*)d_in[7];
  const float* rbb1   = (const float*)d_in[8];
  const float* rw2    = (const float*)d_in[9];
  const float* rb2    = (const float*)d_in[10];
  const float* rg2    = (const float*)d_in[11];
  const float* rbb2   = (const float*)d_in[12];
  const float* ow1    = (const float*)d_in[13];
  const float* ob1    = (const float*)d_in[14];
  const float* og     = (const float*)d_in[15];
  const float* obb    = (const float*)d_in[16];
  const float* ow2    = (const float*)d_in[17];
  const float* ob2    = (const float*)d_in[18];

  char* ws = (char*)d_ws;
  _Float16* A0  = (_Float16*)ws;                       // 67,108,864 B
  size_t offA   = (size_t)NVOX * 128 * sizeof(_Float16);
  _Float16* W1t = (_Float16*)(ws + offA);              // 32 KB
  _Float16* W2t = (_Float16*)(ws + offA + 32768);      // 32 KB
  _Float16* W3t = (_Float16*)(ws + offA + 65536);      // 16 KB

  prep_weights<<<80, 256, 0, stream>>>(rw1, rw2, ow1, W1t, W2t, W3t);
  pool_ln_kernel<<<NVOX, 128, 0, stream>>>(feats, depths, proj, g0, b0, A0);
  mlp_kernel<<<NVOX / 16 / 8, 256, 0, stream>>>(A0,
      W1t, rb1, rg1, rbb1,
      W2t, rb2, rg2, rbb2,
      W3t, ob1, og, obb,
      ow2, ob2, (float*)d_out);
}
